// AutoencoderRVQ_35768487641779
// MI455X (gfx1250) — compile-verified, matrix operand order checked by reading, not run on hardware
//
#include <hip/hip_runtime.h>

typedef __attribute__((ext_vector_type(16))) _Float16 v16h;
typedef __attribute__((ext_vector_type(8)))  _Float16 v8h;
typedef __attribute__((ext_vector_type(8)))  float    v8f;
typedef __attribute__((ext_vector_type(4)))  float    v4f;

#define LEVELS   4
#define K_CODES  4096
#define CDIM     256
#define HW       1024      // H*W
#define NROWS    4096      // B*H*W
#define ROWS_WG  32        // 2 WMMA row-tiles per WG: B fragments reused 2x
#define RSTRIDE  260       // 16B-aligned padded LDS row stride (floats)
#define BETA     0.25f
#define EPSN     1e-12f

// ---------------------------------------------------------------------------
// Prepass: row-normalize codebooks -> f16 (WMMA B operand source), zero qloss.
// One wave per codebook row (L*K = 16384 rows), 8 floats per lane.
// ---------------------------------------------------------------------------
__global__ __launch_bounds__(256) void rvq_prep(const float* __restrict__ cb,
                                                _Float16* __restrict__ cb16,
                                                float* __restrict__ qloss_slot) {
    if (blockIdx.x == 0 && threadIdx.x == 0) *qloss_slot = 0.0f;
    int wave = (blockIdx.x * 256 + threadIdx.x) >> 5;
    int lane = threadIdx.x & 31;
    if (wave >= LEVELS * K_CODES) return;
    const float* row = cb + (size_t)wave * CDIM + lane * 8;
    v4f x0 = *(const v4f*)(row);
    v4f x1 = *(const v4f*)(row + 4);
    float ss = x0[0]*x0[0] + x0[1]*x0[1] + x0[2]*x0[2] + x0[3]*x0[3]
             + x1[0]*x1[0] + x1[1]*x1[1] + x1[2]*x1[2] + x1[3]*x1[3];
    #pragma unroll
    for (int m = 1; m < 32; m <<= 1) ss += __shfl_xor(ss, m, 32);
    float inv = 1.0f / fmaxf(sqrtf(ss), EPSN);
    v8h o;
    #pragma unroll
    for (int j = 0; j < 4; ++j) {
        o[j]     = (_Float16)(x0[j] * inv);
        o[4 + j] = (_Float16)(x1[j] * inv);
    }
    *(v8h*)(cb16 + (size_t)wave * CDIM + lane * 8) = o;
}

// ---------------------------------------------------------------------------
// Main: one WG = 32 rows (two 16-row WMMA tiles), full 4-level RVQ pipeline.
// 8 waves; each wave scans 512 codes per level; every B fragment loaded from
// L2 feeds two v_wmma_f32_16x16x32_f16, with a 2-chunk-deep explicit load
// pipeline so B loads overlap WMMA execution.
// ---------------------------------------------------------------------------
__global__ __launch_bounds__(256) void rvq_main(const float* __restrict__ z,
                                                const float* __restrict__ cb,
                                                const _Float16* __restrict__ cb16,
                                                float* __restrict__ out_q,
                                                float* __restrict__ out_idx,
                                                float* __restrict__ out_qloss) {
    __shared__ float resid[ROWS_WG][RSTRIDE];
    __shared__ float partial[256];
    __shared__ float inv_norm[ROWS_WG];
    __shared__ float cand_val[ROWS_WG][8];
    __shared__ int   cand_idx[ROWS_WG][8];
    __shared__ int   sidx[ROWS_WG];

    const int t     = threadIdx.x;
    const int lane  = t & 31;
    const int wv    = t >> 5;          // 0..7
    const int row_l = t & 31;          // row for load/norm/update phases
    const int cblk  = t >> 5;          // 0..7 (32-channel block per thread)
    const int n0    = blockIdx.x * ROWS_WG;
    const int b     = n0 / HW;
    const int hw0   = n0 % HW;         // 32 | HW, so a WG never straddles b

    // ---- load z (B,C,H,W) -> resid[row][c], coalesced over hw ----
    const float* zb = z + (size_t)b * CDIM * HW + hw0;
    #pragma unroll
    for (int i = 0; i < 32; ++i) {
        int c = cblk + 8 * i;
        resid[row_l][c] = zb[(size_t)c * HW + row_l];
    }

    const int m  = lane & 15;          // WMMA row-in-tile (A) / column (B)
    const int hi = lane >> 4;          // 0: K 0..7/16..23, 1: K 8..15/24..31
    float qacc = 0.0f;                 // sum of squared new-residuals (all levels)

    for (int l = 0; l < LEVELS; ++l) {
        __syncthreads();
        // ---- row norms: thread covers 32 channels of one row ----
        float ps = 0.0f;
        #pragma unroll
        for (int j = 0; j < 32; ++j) {
            float v = resid[row_l][cblk * 32 + j];
            ps += v * v;
        }
        partial[t] = ps;
        __syncthreads();
        if (t < 32) {
            float s = 0.0f;
            #pragma unroll
            for (int j = 0; j < 8; ++j) s += partial[t + 32 * j];
            inv_norm[t] = 1.0f / fmaxf(sqrtf(s), EPSN);
        }
        __syncthreads();

        // ---- build normalized f16 A fragments for both row-tiles ----
        const float inv0 = inv_norm[m];
        const float inv1 = inv_norm[16 + m];
        v16h A0[8], A1[8];
        #pragma unroll
        for (int ci = 0; ci < 8; ++ci) {
            const float* rp0 = &resid[m][ci * 32 + hi * 8];
            const float* rp1 = &resid[16 + m][ci * 32 + hi * 8];
            v4f x0 = *(const v4f*)(rp0);
            v4f x1 = *(const v4f*)(rp0 + 4);
            v4f y0 = *(const v4f*)(rp0 + 16);
            v4f y1 = *(const v4f*)(rp0 + 20);
            v4f u0 = *(const v4f*)(rp1);
            v4f u1 = *(const v4f*)(rp1 + 4);
            v4f w0 = *(const v4f*)(rp1 + 16);
            v4f w1 = *(const v4f*)(rp1 + 20);
            #pragma unroll
            for (int j = 0; j < 4; ++j) {
                A0[ci][j]      = (_Float16)(x0[j] * inv0);
                A0[ci][4 + j]  = (_Float16)(x1[j] * inv0);
                A0[ci][8 + j]  = (_Float16)(y0[j] * inv0);
                A0[ci][12 + j] = (_Float16)(y1[j] * inv0);
                A1[ci][j]      = (_Float16)(u0[j] * inv1);
                A1[ci][4 + j]  = (_Float16)(u1[j] * inv1);
                A1[ci][8 + j]  = (_Float16)(w0[j] * inv1);
                A1[ci][12 + j] = (_Float16)(w1[j] * inv1);
            }
        }

        // ---- scan this wave's 512 codes: 32 tiles x (8 chunks x 2 WMMAs) ----
        float best0[8], best1[8];
        int   bidx0[8], bidx1[8];
        #pragma unroll
        for (int r = 0; r < 8; ++r) {
            best0[r] = -3.0e38f; bidx0[r] = 0;
            best1[r] = -3.0e38f; bidx1[r] = 0;
        }

        const _Float16* cbl = cb16 + (size_t)l * K_CODES * CDIM;
        int code = (wv << 9) + m;                       // wave base + lane column
        for (int kt = 0; kt < 32; ++kt, code += 16) {
            const _Float16* brow = cbl + (size_t)code * CDIM + hi * 8;
            __builtin_prefetch((const void*)(brow + 16 * CDIM), 0, 0);

            // 2-chunk-deep explicit B pipeline: issue loads for chunk ci+2
            // before the WMMAs of chunk ci so L2 latency overlaps compute.
            v8h pb0 = *(const v8h*)(brow);
            v8h pb1 = *(const v8h*)(brow + 16);
            v8h nb0 = *(const v8h*)(brow + 32);
            v8h nb1 = *(const v8h*)(brow + 48);
            v8f acc0 = {}, acc1 = {};
            #pragma unroll
            for (int ci = 0; ci < 8; ++ci) {
                v16h Bv;
                #pragma unroll
                for (int j = 0; j < 8; ++j) { Bv[j] = pb0[j]; Bv[8 + j] = pb1[j]; }
                pb0 = nb0;
                pb1 = nb1;
                if (ci < 6) {
                    nb0 = *(const v8h*)(brow + (ci + 2) * 32);
                    nb1 = *(const v8h*)(brow + (ci + 2) * 32 + 16);
                }
                acc0 = __builtin_amdgcn_wmma_f32_16x16x32_f16(
                         false, A0[ci], false, Bv, (short)0, acc0, false, false);
                acc1 = __builtin_amdgcn_wmma_f32_16x16x32_f16(
                         false, A1[ci], false, Bv, (short)0, acc1, false, false);
            }
            #pragma unroll
            for (int r = 0; r < 8; ++r) {
                if (acc0[r] > best0[r]) { best0[r] = acc0[r]; bidx0[r] = code; }
                if (acc1[r] > best1[r]) { best1[r] = acc1[r]; bidx1[r] = code; }
            }
        }

        // ---- argmax reduce across the 16 lanes of each half ----
        #pragma unroll
        for (int r = 0; r < 8; ++r) {
            float v0 = best0[r]; int i0 = bidx0[r];
            float v1 = best1[r]; int i1 = bidx1[r];
            #pragma unroll
            for (int msk = 1; msk < 16; msk <<= 1) {
                float ov0 = __shfl_xor(v0, msk, 32);
                int   oi0 = __shfl_xor(i0, msk, 32);
                float ov1 = __shfl_xor(v1, msk, 32);
                int   oi1 = __shfl_xor(i1, msk, 32);
                if (ov0 > v0 || (ov0 == v0 && oi0 < i0)) { v0 = ov0; i0 = oi0; }
                if (ov1 > v1 || (ov1 == v1 && oi1 < i1)) { v1 = ov1; i1 = oi1; }
            }
            best0[r] = v0; bidx0[r] = i0;
            best1[r] = v1; bidx1[r] = i1;
        }
        if (m == 0) {
            int rbase = hi * 8;                 // rows r..r+7 of each tile
            #pragma unroll
            for (int r = 0; r < 8; ++r) {
                cand_val[rbase + r][wv]      = best0[r];
                cand_idx[rbase + r][wv]      = bidx0[r];
                cand_val[16 + rbase + r][wv] = best1[r];
                cand_idx[16 + rbase + r][wv] = bidx1[r];
            }
        }
        __syncthreads();
        if (t < 32) {
            float bv = cand_val[t][0];
            int   bi = cand_idx[t][0];
            #pragma unroll
            for (int w2 = 1; w2 < 8; ++w2) {
                float v2 = cand_val[t][w2];
                int   i2 = cand_idx[t][w2];
                if (v2 > bv || (v2 == bv && i2 < bi)) { bv = v2; bi = i2; }
            }
            sidx[t] = bi;
            out_idx[(size_t)l * NROWS + n0 + t] = (float)bi;
        }
        __syncthreads();

        // ---- residual update (fp32 gather from raw codebook) + qloss accum ----
        const float* zq = cb + ((size_t)l * K_CODES + sidx[row_l]) * CDIM;
        #pragma unroll
        for (int j = 0; j < 32; ++j) {
            int c = cblk * 32 + j;
            float nr = resid[row_l][c] - zq[c];
            resid[row_l][c] = nr;
            qacc += nr * nr;            // (z_q - residual)^2 == new_residual^2
        }
    }

    __syncthreads();
    // ---- q = z - residual_final, written back in (B,C,H,W) layout ----
    float* ob = out_q + (size_t)b * CDIM * HW + hw0;
    #pragma unroll
    for (int i = 0; i < 32; ++i) {
        int c = cblk + 8 * i;
        size_t a = (size_t)c * HW + row_l;
        ob[a] = zb[a] - resid[row_l][c];
    }

    // ---- qloss reduction: wave shuffle -> LDS -> one atomicAdd per WG ----
    #pragma unroll
    for (int msk = 1; msk < 32; msk <<= 1) qacc += __shfl_xor(qacc, msk, 32);
    if (lane == 0) partial[wv] = qacc;
    __syncthreads();
    if (t == 0) {
        float s = 0.0f;
        #pragma unroll
        for (int w2 = 0; w2 < 8; ++w2) s += partial[w2];
        atomicAdd(out_qloss, s * (BETA / (float)((size_t)NROWS * CDIM)));
    }
}

// ---------------------------------------------------------------------------
extern "C" void kernel_launch(void* const* d_in, const int* in_sizes, int n_in,
                              void* d_out, int out_size, void* d_ws, size_t ws_size,
                              hipStream_t stream) {
    (void)in_sizes; (void)n_in; (void)out_size; (void)ws_size;
    const float* z  = (const float*)d_in[0];           // (4,256,32,32)
    const float* cb = (const float*)d_in[1];           // (4,4096,256)
    _Float16* cb16  = (_Float16*)d_ws;                 // 4*4096*256 f16 = 8 MB

    float* out   = (float*)d_out;
    float* out_q = out;                                // 1048576 floats (B,C,H,W)
    float* out_i = out + (size_t)NROWS * CDIM;         // 16384 floats (L,B,H,W)
    float* out_l = out_i + (size_t)LEVELS * NROWS;     // 1 float qloss

    // normalize codebooks -> f16 (one wave per row) and zero the qloss slot
    rvq_prep<<<(LEVELS * K_CODES * 32) / 256, 256, 0, stream>>>(cb, cb16, out_l);
    // 128 WGs x 32 rows: full 4-level RVQ per workgroup, 2x B reuse
    rvq_main<<<NROWS / ROWS_WG, 256, 0, stream>>>(z, cb, cb16, out_q, out_i, out_l);
}
